// TopKAttention_80530636800342
// MI455X (gfx1250) — matrix operand order repaired and checked by reading, not compiled, not run on hardware
//
#include <hip/hip_runtime.h>
#include <hip/hip_bf16.h>

#define B_    2
#define S_    2048
#define D_    1024
#define H_    16
#define DH_   64
#define KTOP_ 64
#define SCALE_ 0.125f   // DH^-0.5

typedef __attribute__((ext_vector_type(16))) __bf16 v16bf;
typedef __attribute__((ext_vector_type(8)))  __bf16 v8bf;
typedef __attribute__((ext_vector_type(8)))  float  v8f;
typedef __attribute__((ext_vector_type(4)))  unsigned uint32x4;
typedef __attribute__((ext_vector_type(8)))  int      int32x8;
typedef __attribute__((ext_vector_type(4)))  int      int32x4;

__device__ __forceinline__ v8bf cvt8(const float* p) {
  v8bf r;
  #pragma unroll
  for (int i = 0; i < 8; ++i) r[i] = (__bf16)p[i];
  return r;
}

__device__ __forceinline__ v16bf join16(v8bf lo, v8bf hi) {
  return __builtin_shufflevector(lo, hi, 0,1,2,3,4,5,6,7,8,9,10,11,12,13,14,15);
}

// monotonic float->uint key (larger float -> larger uint)
__device__ __forceinline__ unsigned orderKey(float f) {
  unsigned u = __float_as_uint(f);
  return u ^ ((u >> 31) ? 0xFFFFFFFFu : 0x80000000u);
}

// TDM: async-copy a [16 keys x 64 dims] bf16 tile (row-major, stride 64) from
// global to LDS. D# per CDNA5 ISA (Group0/Group1 bitfields).
__device__ __forceinline__ void tdm_load_tile(unsigned lds_off, const __bf16* gaddr) {
  unsigned long long ga = (unsigned long long)(size_t)gaddr;
  uint32x4 g0;
  g0[0] = 1u;                                            // count=1, user desc
  g0[1] = lds_off;                                       // lds_addr (bytes)
  g0[2] = (unsigned)(ga & 0xFFFFFFFFu);                  // global_addr[31:0]
  g0[3] = (unsigned)((ga >> 32) & 0x01FFFFFFu) | (2u << 30); // [56:32] | type=2
  int32x8 g1;
  g1[0] = (int)(1u << 16);        // workgroup_mask=0, data_size=1 (2 bytes)
  g1[1] = (int)(64u << 16);       // tensor_dim0 = 64  (bits 63:48)
  g1[2] = (int)(16u << 16);       // tensor_dim1 = 16  (bits 95:80)
  g1[3] = (int)(64u << 16);       // tile_dim0  = 64   (bits 127:112)
  g1[4] = 16;                     // tile_dim1  = 16   (bits 143:128)
  g1[5] = 64;                     // tensor_dim0_stride = 64 (bits 191:160)
  g1[6] = 0;
  g1[7] = 0;
  int32x4 z4 = {0, 0, 0, 0};
  int32x8 z8 = {0, 0, 0, 0, 0, 0, 0, 0};
  __builtin_amdgcn_tensor_load_to_lds(g0, g1, z4, z4, z8, 0);
}

// ---------------------------------------------------------------------------
// Kernel 1: Q = query @ Wq^T, K = key @ Wk^T, bf16 out in [B,H,S,DH].
// One wave computes a 32x32 output tile (2x2 WMMA fragments): 4 WMMAs per
// fragment-set, 2x A/B reuse vs a 16x16 tile. K-loop in chunks of 32 with
// v_wmma_f32_16x16x32_bf16; f32->bf16 conversion inline (v_cvt_pk_bf16_f32
// co-issues with WMMA per round-1 asm).
// ---------------------------------------------------------------------------
__global__ void __launch_bounds__(256)
proj_kernel(const float* __restrict__ query, const float* __restrict__ key,
            const float* __restrict__ Wq,    const float* __restrict__ Wk,
            __bf16* __restrict__ Qb,         __bf16* __restrict__ Kb) {
  const int lane = threadIdx.x & 31;
  const int wid  = blockIdx.x * (blockDim.x >> 5) + (threadIdx.x >> 5);
  const int half = lane >> 4;
  const int r    = lane & 15;

  const int tiles_per_mat = (B_ * S_ / 32) * (D_ / 32);   // 128*32 = 4096
  const int which = wid / tiles_per_mat;                  // 0=Q, 1=K
  const int t     = wid % tiles_per_mat;
  const int rt    = t / (D_ / 32);
  const int ct    = t % (D_ / 32);

  const float* X   = which ? key : query;
  const float* W   = which ? Wk  : Wq;
  __bf16*      Out = which ? Kb  : Qb;

  const float* xrow0 = X + (size_t)(rt * 32 + r) * D_;
  const float* xrow1 = xrow0 + (size_t)16 * D_;
  const float* wrow0 = W + (size_t)(ct * 32 + r) * D_;
  const float* wrow1 = wrow0 + (size_t)16 * D_;

  v8f c00 = {}, c01 = {}, c10 = {}, c11 = {};
  #pragma unroll 2
  for (int kk = 0; kk < D_ / 32; ++kk) {
    const int kb = kk * 32;
    v16bf a0 = join16(cvt8(xrow0 + kb + half * 8), cvt8(xrow0 + kb + 16 + half * 8));
    v16bf a1 = join16(cvt8(xrow1 + kb + half * 8), cvt8(xrow1 + kb + 16 + half * 8));
    v16bf b0 = join16(cvt8(wrow0 + kb + half * 8), cvt8(wrow0 + kb + 16 + half * 8));
    v16bf b1 = join16(cvt8(wrow1 + kb + half * 8), cvt8(wrow1 + kb + 16 + half * 8));
    c00 = __builtin_amdgcn_wmma_f32_16x16x32_bf16(false, a0, false, b0, (short)0, c00, false, false);
    c01 = __builtin_amdgcn_wmma_f32_16x16x32_bf16(false, a0, false, b1, (short)0, c01, false, false);
    c10 = __builtin_amdgcn_wmma_f32_16x16x32_bf16(false, a1, false, b0, (short)0, c10, false, false);
    c11 = __builtin_amdgcn_wmma_f32_16x16x32_bf16(false, a1, false, b1, (short)0, c11, false, false);
  }

  #pragma unroll
  for (int rf = 0; rf < 2; ++rf) {
    #pragma unroll
    for (int cf = 0; cf < 2; ++cf) {
      const v8f c = (rf == 0) ? (cf == 0 ? c00 : c01) : (cf == 0 ? c10 : c11);
      const int cc0 = ct * 32 + cf * 16;
      const int h   = cc0 >> 6;
      const int dh0 = cc0 & 63;
      #pragma unroll
      for (int v = 0; v < 8; ++v) {
        const int R = rt * 32 + rf * 16 + half * 8 + v;  // C layout: M=8*half+v, N=r
        const int b = R >> 11, s = R & (S_ - 1);
        Out[((size_t)((b * H_ + h) * S_ + s)) * DH_ + dh0 + r] = (__bf16)c[v];
      }
    }
  }
}

// ---------------------------------------------------------------------------
// Kernel 2: fused scores + softmax + top-64 + sparse V contraction.
// Block = 256 threads (8 waves) per (b, h, 16-query tile).
// Phase 1: per-wave TDM double-buffer streams K tiles into LDS
//          (tensor_load_to_lds + s_wait_tensorcnt); WMMA score strip
//          [16 x 2048] parked in LDS (never touches HBM).
// Phase 2: per-row radix-select of 64th-largest score via ds_add_u32
//          histograms, exact softmax stats, ballot-driven sparse V gather.
// LDS: 128K scores + 32K K double-buffers + 8K hist = 168K (<320K WGP LDS).
// ---------------------------------------------------------------------------
#define SCORES_BYTES (16 * S_ * 4)
#define KBUF_BYTES   2048
#define KBUF_BASE    SCORES_BYTES
#define HIST_BASE    (KBUF_BASE + 8 * 2 * KBUF_BYTES)
#define SMEM_TOTAL   (HIST_BASE + 8 * 256 * 4)

__global__ void __launch_bounds__(256)
attn_kernel(const __bf16* __restrict__ Qb, const __bf16* __restrict__ Kb,
            const float* __restrict__ value, float* __restrict__ out) {
  extern __shared__ char smem_raw[];
  float*    scores = (float*)smem_raw;
  unsigned* hist   = (unsigned*)(smem_raw + HIST_BASE);

  const int lane = threadIdx.x & 31;
  const int wave = threadIdx.x >> 5;     // 0..7
  const int half = lane >> 4;
  const int r    = lane & 15;

  const int bh = blockIdx.x / (S_ / 16);
  const int qt = blockIdx.x % (S_ / 16);
  const int b  = bh / H_, h = bh % H_;

  const __bf16* qbase = Qb + ((size_t)bh * S_ + qt * 16) * DH_;
  const __bf16* kbase = Kb + (size_t)bh * S_ * DH_;

  // ---- Phase 1: Q-tile A fragments (16x64 = two 16x32 chunks) ----
  const __bf16* qrow = qbase + (size_t)r * DH_;
  v16bf a0 = join16(*(const v8bf*)(qrow + half * 8),
                    *(const v8bf*)(qrow + 16 + half * 8));
  v16bf a1 = join16(*(const v8bf*)(qrow + 32 + half * 8),
                    *(const v8bf*)(qrow + 48 + half * 8));

  // double-buffered TDM stream of this wave's 16 K-tiles (keys wave*256..)
  char* kbuf[2] = { smem_raw + KBUF_BASE + (wave * 2 + 0) * KBUF_BYTES,
                    smem_raw + KBUF_BASE + (wave * 2 + 1) * KBUF_BYTES };
  const unsigned koff[2] = { (unsigned)(size_t)(void*)kbuf[0],
                             (unsigned)(size_t)(void*)kbuf[1] };
  const int kt0 = wave * 16;

  auto consume_tile = [&](int i) {
    const int kt = kt0 + i;
    const __bf16* krow = (const __bf16*)kbuf[i & 1] + (size_t)r * DH_;
    v16bf b0 = join16(*(const v8bf*)(krow + half * 8), *(const v8bf*)(krow + 16 + half * 8));        // ds_load
    v16bf b1 = join16(*(const v8bf*)(krow + 32 + half * 8), *(const v8bf*)(krow + 48 + half * 8));
    v8f c = {};
    c = __builtin_amdgcn_wmma_f32_16x16x32_bf16(false, a0, false, b0, (short)0, c, false, false);
    c = __builtin_amdgcn_wmma_f32_16x16x32_bf16(false, a1, false, b1, (short)0, c, false, false);
    #pragma unroll
    for (int v = 0; v < 8; ++v)
      scores[(half * 8 + v) * S_ + kt * 16 + r] = c[v] * SCALE_;
  };

  tdm_load_tile(koff[0], kbase + (size_t)kt0 * 16 * DH_);
  for (int i = 0; i < 15; ++i) {
    tdm_load_tile(koff[(i + 1) & 1], kbase + (size_t)(kt0 + i + 1) * 16 * DH_);
    __builtin_amdgcn_s_wait_tensorcnt(1);   // buffer i ready, next still in flight
    consume_tile(i);
  }
  __builtin_amdgcn_s_wait_tensorcnt(0);     // last buffer ready
  consume_tile(15);

  __syncthreads();

  // ---- Phase 2: one wave per row; wave handles rows {wave, wave+8} ----
  unsigned* myhist = hist + wave * 256;
  for (int rowSel = 0; rowSel < 2; ++rowSel) {
    const int row = wave + rowSel * 8;
    const float* srow = scores + (size_t)row * S_;

    // row max
    float m = -3.4e38f;
    for (int it = 0; it < S_ / 32; ++it) m = fmaxf(m, srow[it * 32 + lane]);
    #pragma unroll
    for (int off = 16; off; off >>= 1) m = fmaxf(m, __shfl_xor(m, off, 32));

    // radix select: exact bit pattern of the 64th-largest score
    unsigned prefix = 0, pmask = 0;
    int remaining = KTOP_;
    #pragma unroll 1
    for (int pass = 0; pass < 4; ++pass) {
      const int shift = 24 - 8 * pass;
      #pragma unroll
      for (int tt = 0; tt < 8; ++tt) myhist[lane * 8 + tt] = 0u;
      __threadfence_block();
      for (int it = 0; it < S_ / 32; ++it) {
        unsigned u = orderKey(srow[it * 32 + lane]);
        if ((u & pmask) == prefix)
          atomicAdd(&myhist[(u >> shift) & 255], 1u);   // ds_add_u32
      }
      __threadfence_block();
      unsigned cnt[8]; int lsum = 0;
      #pragma unroll
      for (int tt = 0; tt < 8; ++tt) { cnt[tt] = myhist[lane * 8 + tt]; lsum += (int)cnt[tt]; }
      // suffix sum over lanes: x = count in my digits and above
      int x = lsum;
      #pragma unroll
      for (int off = 1; off < 32; off <<= 1) {
        int y = __shfl_down(x, off, 32);
        if (lane + off < 32) x += y;
      }
      int g = x - lsum;                 // count strictly above my digit range
      int foundDigit = -1, newRem = 0;
      #pragma unroll
      for (int tt = 7; tt >= 0; --tt) {
        int cc = (int)cnt[tt];
        if (foundDigit < 0 && remaining > g && remaining <= g + cc) {
          foundDigit = lane * 8 + tt; newRem = remaining - g;
        }
        g += cc;
      }
      unsigned fmask = (unsigned)__ballot(foundDigit >= 0);
      int src = __ffs(fmask) - 1;
      int digit = __shfl(foundDigit, src, 32);
      remaining = __shfl(newRem, src, 32);
      prefix |= (unsigned)digit << shift;
      pmask  |= 0xFFu << shift;
    }
    const unsigned thr = prefix;

    // softmax stats: full Z and top-k sum E  =>  w = exp/(E + 1e-9*Z)
    float Z = 0.f, E = 0.f;
    for (int it = 0; it < S_ / 32; ++it) {
      float s = srow[it * 32 + lane];
      float e = __expf(s - m);
      Z += e;
      if (orderKey(s) >= thr) E += e;
    }
    #pragma unroll
    for (int off = 16; off; off >>= 1) {
      Z += __shfl_xor(Z, off, 32);
      E += __shfl_xor(E, off, 32);
    }
    const float inv = 1.0f / (E + 1e-9f * Z);

    // sparse weighted V gather: lane owns dims {2*lane, 2*lane+1}
    float acc0 = 0.f, acc1 = 0.f;
    const float* vbase = value + (size_t)b * S_ * D_ + h * DH_ + lane * 2;
    for (int c = 0; c < S_ / 32; ++c) {
      float s = srow[c * 32 + lane];
      bool sel = orderKey(s) >= thr;
      float w = sel ? __expf(s - m) * inv : 0.f;
      unsigned mask = (unsigned)__ballot(sel);
      while (mask) {
        int j = __ffs(mask) - 1; mask &= mask - 1;
        float wj = __shfl(w, j, 32);
        const float2 vv = *(const float2*)(vbase + (size_t)(c * 32 + j) * D_);
        acc0 += wj * vv.x; acc1 += wj * vv.y;
      }
    }
    const int qrow = qt * 16 + row;
    float* op = out + ((size_t)b * S_ + qrow) * D_ + h * DH_ + lane * 2;
    op[0] = acc0; op[1] = acc1;
  }
}

extern "C" void kernel_launch(void* const* d_in, const int* in_sizes, int n_in,
                              void* d_out, int out_size, void* d_ws, size_t ws_size,
                              hipStream_t stream) {
  (void)in_sizes; (void)n_in; (void)out_size; (void)ws_size;
  const float* query = (const float*)d_in[0];
  const float* key   = (const float*)d_in[1];
  const float* value = (const float*)d_in[2];
  const float* Wq    = (const float*)d_in[3];
  const float* Wk    = (const float*)d_in[4];
  float* out = (float*)d_out;

  __bf16* Qb = (__bf16*)d_ws;                       // 8 MB
  __bf16* Kb = Qb + (size_t)B_ * H_ * S_ * DH_;     // 8 MB

  dim3 blk(256);
  const int projBlocks = 2 * (B_ * S_ / 32) * (D_ / 32) / 8;   // 1024
  proj_kernel<<<projBlocks, blk, 0, stream>>>(query, key, Wq, Wk, Qb, Kb);

  const int attnBlocks = B_ * H_ * (S_ / 16);                  // 4096
  attn_kernel<<<attnBlocks, blk, SMEM_TOTAL, stream>>>(Qb, Kb, value, out);
}
